// SessionGraph_15229954031622
// MI455X (gfx1250) — matrix operand order repaired and checked by reading, compiled, not run on hardware
//
#include <hip/hip_runtime.h>
#include <hip/hip_bf16.h>

typedef __attribute__((ext_vector_type(16))) _Float16 v16h;
typedef __attribute__((ext_vector_type(8)))  _Float16 v8h;
typedef __attribute__((ext_vector_type(8)))  float    v8f;

#define ALPHA 0.2f
#define NEGV  -9e15f

// dims
#define BB 512
#define LL 64
#define NEE 128
#define DD 128
#define BLD  ((size_t)BB * LL * DD)    // 4,194,304
#define BNED ((size_t)BB * NEE * DD)   // 8,388,608

#define WSTRIDE 136                    // padded f16 row stride (272B, bank-skewed)

// ---------------------------------------------------------------------------
// Embedding gather: resid = emb[inputs], nodes2_out = emb2[inputs]
// ---------------------------------------------------------------------------
__global__ __launch_bounds__(256) void k_gather(
    const int* __restrict__ idx,
    const float* __restrict__ emb, const float* __restrict__ emb2,
    float* __restrict__ resid, float* __restrict__ nodes2)
{
    int row  = blockIdx.x * 8 + (threadIdx.x >> 5);
    int lane = threadIdx.x & 31;
    int e = idx[row];
    const float4* s1 = (const float4*)(emb  + (size_t)e * DD);
    const float4* s2 = (const float4*)(emb2 + (size_t)e * DD);
    ((float4*)(resid  + (size_t)row * DD))[lane] = s1[lane];
    ((float4*)(nodes2 + (size_t)row * DD))[lane] = s2[lane];
}

// ---------------------------------------------------------------------------
// c = sum_k q[k]*a[k], k<128  (one warp)
// ---------------------------------------------------------------------------
__global__ void k_dot128(const float* __restrict__ q,
                         const float* __restrict__ a,
                         float* __restrict__ out)
{
    int lane = threadIdx.x & 31;
    float4 qa = ((const float4*)q)[lane];
    float4 av = ((const float4*)a)[lane];
    float s = qa.x*av.x + qa.y*av.y + qa.z*av.z + qa.w*av.w;
    for (int m = 16; m >= 1; m >>= 1) s += __shfl_xor(s, m, 32);
    if (lane == 0) out[0] = s;
}

// ---------------------------------------------------------------------------
// WMMA GEMM: Y[M,128] = X[M,128] @ W[128,128]   (f32 io, f16 WMMA, f32 acc)
// block = 256 thr = 8 waves; block owns 64 rows (4 m-tiles), wave owns n-tile.
// W staged TRANSPOSED (fragment-major) + X tile staged, both f16 in LDS, so
// every WMMA fragment is two contiguous ds_load_b128's.
//   B frag (lane): col n = wtile*16 + (lane&15); K = kb*32 + (lane>=16?16:0)+e
//                  -> Wt[n][ks .. ks+15]   (contiguous)
//   A frag (lane): row = mtile*16 + (lane&15); K = kb*32 + (lane>=16?16:0)+e
//                  -> Xl[row][ks .. ks+15]   (contiguous)
//   C/D: col = lane&15; vgpr vr -> row vr + (lane>=16 ? 8 : 0).
// ---------------------------------------------------------------------------
__global__ __launch_bounds__(256) void k_gemm128(
    const float* __restrict__ X, const float* __restrict__ W,
    float* __restrict__ Y, int M)
{
    __shared__ __attribute__((aligned(16))) _Float16 Wt[128 * WSTRIDE]; // [n][k]
    __shared__ __attribute__((aligned(16))) _Float16 Xl[64  * WSTRIDE]; // [row][k]
    int tid = threadIdx.x;
    int m0  = blockIdx.x * 64;

    // stage W transposed as packed f16 pairs (one ds_store_b32 per pair)
    for (int t = tid; t < 128 * 64; t += 256) {
        int n = t & 127, k2 = t >> 7;                 // k2 = K pair index
        union { _Float16 h[2]; unsigned u; } p;
        p.h[0] = (_Float16)W[(2 * k2)     * DD + n];
        p.h[1] = (_Float16)W[(2 * k2 + 1) * DD + n];
        *reinterpret_cast<unsigned*>(&Wt[n * WSTRIDE + 2 * k2]) = p.u;
    }
    // stage X rows m0..m0+63 as packed f16 pairs
    const float* Xb = X + (size_t)m0 * DD;
    for (int t = tid; t < 64 * 64; t += 256) {
        int k2 = t & 63, row = t >> 6;
        float2 xv = *reinterpret_cast<const float2*>(&Xb[(size_t)row * DD + 2 * k2]);
        union { _Float16 h[2]; unsigned u; } p;
        p.h[0] = (_Float16)xv.x; p.h[1] = (_Float16)xv.y;
        *reinterpret_cast<unsigned*>(&Xl[row * WSTRIDE + 2 * k2]) = p.u;
    }
    __syncthreads();

    int wave = tid >> 5, lane = tid & 31;
    int r = lane & 15, half = lane >> 4;

    // hoisted B fragments (reused across 4 m-tiles)
    v16h bfr[4];
#pragma unroll
    for (int kb = 0; kb < 4; ++kb) {
        const _Float16* p = &Wt[(wave * 16 + r) * WSTRIDE + kb * 32 + half * 16];
        v8h lo = *reinterpret_cast<const v8h*>(p);
        v8h hi = *reinterpret_cast<const v8h*>(p + 8);
        bfr[kb] = __builtin_shufflevector(lo, hi,
                     0,1,2,3,4,5,6,7,8,9,10,11,12,13,14,15);
    }

    int n = wave * 16 + r;
#pragma unroll
    for (int mt = 0; mt < 4; ++mt) {
        v8f acc = {};
#pragma unroll
        for (int kb = 0; kb < 4; ++kb) {
            const _Float16* q = &Xl[(mt * 16 + r) * WSTRIDE + kb * 32 + half * 16];
            v8h lo = *reinterpret_cast<const v8h*>(q);       // K ks..ks+7
            v8h hi = *reinterpret_cast<const v8h*>(q + 8);   // K ks+8..ks+15
            v16h a = __builtin_shufflevector(lo, hi,
                        0,1,2,3,4,5,6,7,8,9,10,11,12,13,14,15);
            acc = __builtin_amdgcn_wmma_f32_16x16x32_f16(
                false, a, false, bfr[kb], (short)0, acc, false, false);
        }
        float* yout = Y + (size_t)(m0 + mt * 16) * DD + n;
#pragma unroll
        for (int vr = 0; vr < 8; ++vr)
            yout[(size_t)(vr + half * 8) * DD] = acc[vr];
    }
    (void)M;
}

// ---------------------------------------------------------------------------
// rowdot: out[m] = act( c + sum_k X[m,k]*v[k] ), one warp per row
// ---------------------------------------------------------------------------
__global__ __launch_bounds__(256) void k_rowdot(
    const float* __restrict__ X, const float* __restrict__ v,
    const float* __restrict__ cscalar, float* __restrict__ out,
    int M, int lrelu)
{
    int row  = blockIdx.x * 8 + (threadIdx.x >> 5);
    int lane = threadIdx.x & 31;
    if (row >= M) return;
    float4 a  = ((const float4*)(X + (size_t)row * DD))[lane];
    float4 bv = ((const float4*)v)[lane];
    float s = a.x*bv.x + a.y*bv.y + a.z*bv.z + a.w*bv.w;
    for (int m = 16; m >= 1; m >>= 1) s += __shfl_xor(s, m, 32);
    if (lane == 0) {
        float r = s + (cscalar ? cscalar[0] : 0.f);
        if (lrelu) r = r > 0.f ? r : ALPHA * r;
        out[row] = r;
    }
}

// ---------------------------------------------------------------------------
// node->edge attention + aggregation:
//   att[i,j] = softmax_j( adj[b,i,j]>0 ? e1[b,j] : NEG )
//   edge[b,i,:] = sum_j att[i,j] * xt[b,j,:]
// ---------------------------------------------------------------------------
__global__ __launch_bounds__(256) void k_edge_agg(
    const float* __restrict__ xt, const float* __restrict__ adj,
    const float* __restrict__ e1, float* __restrict__ edge)
{
    __shared__ float xts[LL * DD];   // 32 KB
    __shared__ float e1s[LL];
    __shared__ float attw[8][LL];
    int b = blockIdx.x, tid = threadIdx.x;
    const float* xtb = xt + (size_t)b * LL * DD;
    for (int t = tid; t < LL * DD; t += 256) xts[t] = xtb[t];
    if (tid < LL) e1s[tid] = e1[b * LL + tid];
    __syncthreads();

    int wave = tid >> 5, lane = tid & 31;
    int i0 = blockIdx.y * 32 + wave * 4;
    const float* adjb = adj + (size_t)b * NEE * LL;
    for (int ii = 0; ii < 4; ++ii) {
        int i = i0 + ii;
        const float* arow = adjb + (size_t)i * LL;
        float l0 = arow[lane]      > 0.f ? e1s[lane]      : NEGV;
        float l1 = arow[lane + 32] > 0.f ? e1s[lane + 32] : NEGV;
        float mx = fmaxf(l0, l1);
        for (int m = 16; m >= 1; m >>= 1) mx = fmaxf(mx, __shfl_xor(mx, m, 32));
        float w0 = __expf(l0 - mx), w1 = __expf(l1 - mx);
        float s = w0 + w1;
        for (int m = 16; m >= 1; m >>= 1) s += __shfl_xor(s, m, 32);
        float inv = 1.f / s;
        attw[wave][lane]      = w0 * inv;
        attw[wave][lane + 32] = w1 * inv;
        __syncthreads();
        float a0 = 0, a1 = 0, a2 = 0, a3 = 0;
        for (int j = 0; j < LL; ++j) {
            float aw = attw[wave][j];
            const float* xr = &xts[j * DD];
            a0 += aw * xr[lane];
            a1 += aw * xr[lane + 32];
            a2 += aw * xr[lane + 64];
            a3 += aw * xr[lane + 96];
        }
        float* eo = edge + ((size_t)b * NEE + i) * DD;
        eo[lane] = a0; eo[lane + 32] = a1; eo[lane + 64] = a2; eo[lane + 96] = a3;
    }
}

// ---------------------------------------------------------------------------
// edge->node attention + aggregation + residual:
//   lg[i,j] = adj>0 ? lrelu(a2x[j]+a2e[i]) : NEG ; softmax over i
//   out[b,j,:] = sum_i att[i] * edge[b,i,:] + resid[b,j,:]
// ---------------------------------------------------------------------------
__global__ __launch_bounds__(256) void k_node_agg(
    const float* __restrict__ edge, const float* __restrict__ adj,
    const float* __restrict__ a2x, const float* __restrict__ a2e,
    const float* __restrict__ resid,
    float* __restrict__ out1, float* __restrict__ out2)
{
    __shared__ float es[NEE * DD];   // 64 KB
    __shared__ float adjs[NEE * LL]; // 32 KB
    __shared__ float a2es[NEE];
    __shared__ float a2xs[LL];
    __shared__ float attw[8][NEE];
    int b = blockIdx.x, tid = threadIdx.x;
    const float* eb = edge + (size_t)b * NEE * DD;
    for (int t = tid; t < NEE * DD; t += 256) es[t] = eb[t];
    const float* ab = adj + (size_t)b * NEE * LL;
    for (int t = tid; t < NEE * LL; t += 256) adjs[t] = ab[t];
    if (tid < NEE) a2es[tid] = a2e[b * NEE + tid];
    if (tid < LL)  a2xs[tid] = a2x[b * LL + tid];
    __syncthreads();

    int wave = tid >> 5, lane = tid & 31;
    int j0 = blockIdx.y * 32 + wave * 4;
    for (int jj = 0; jj < 4; ++jj) {
        int j = j0 + jj;
        float ax = a2xs[j];
        float lg[4], mx = NEGV;
#pragma unroll
        for (int c = 0; c < 4; ++c) {
            int i = lane + c * 32;
            float v = ax + a2es[i];
            v = v > 0.f ? v : ALPHA * v;
            lg[c] = adjs[i * LL + j] > 0.f ? v : NEGV;
            mx = fmaxf(mx, lg[c]);
        }
        for (int m = 16; m >= 1; m >>= 1) mx = fmaxf(mx, __shfl_xor(mx, m, 32));
        float s = 0.f, w[4];
#pragma unroll
        for (int c = 0; c < 4; ++c) { w[c] = __expf(lg[c] - mx); s += w[c]; }
        for (int m = 16; m >= 1; m >>= 1) s += __shfl_xor(s, m, 32);
        float inv = 1.f / s;
#pragma unroll
        for (int c = 0; c < 4; ++c) attw[wave][lane + c * 32] = w[c] * inv;
        __syncthreads();
        float a0 = 0, a1 = 0, a2v = 0, a3 = 0;
        for (int i = 0; i < NEE; ++i) {
            float aw = attw[wave][i];
            const float* er = &es[i * DD];
            a0  += aw * er[lane];
            a1  += aw * er[lane + 32];
            a2v += aw * er[lane + 64];
            a3  += aw * er[lane + 96];
        }
        size_t off = ((size_t)b * LL + j) * DD;
        const float* rr = resid + off;
        float o0 = a0  + rr[lane],      o1 = a1 + rr[lane + 32];
        float o2 = a2v + rr[lane + 64], o3 = a3 + rr[lane + 96];
        out1[off + lane] = o0; out1[off + lane + 32] = o1;
        out1[off + lane + 64] = o2; out1[off + lane + 96] = o3;
        if (out2) {
            out2[off + lane] = o0; out2[off + lane + 32] = o1;
            out2[off + lane + 64] = o2; out2[off + lane + 96] = o3;
        }
    }
}

// ---------------------------------------------------------------------------
extern "C" void kernel_launch(void* const* d_in, const int* in_sizes, int n_in,
                              void* d_out, int out_size, void* d_ws, size_t ws_size,
                              hipStream_t stream) {
    (void)in_sizes; (void)n_in; (void)out_size; (void)ws_size;

    const int*   inputs = (const int*)d_in[0];
    const float* HT     = (const float*)d_in[1];
    const float* emb    = (const float*)d_in[4];
    const float* emb2   = (const float*)d_in[5];
    const float* g1_w2  = (const float*)d_in[6];
    const float* g1_w3  = (const float*)d_in[7];
    const float* g1_q   = (const float*)d_in[8];
    const float* g1_a   = (const float*)d_in[9];
    const float* g1_a2  = (const float*)d_in[10];
    const float* g2_w   = (const float*)d_in[11];
    const float* g2_w2  = (const float*)d_in[12];
    const float* g2_w3  = (const float*)d_in[13];
    const float* g2_q   = (const float*)d_in[14];
    const float* g2_a   = (const float*)d_in[15];
    const float* g2_a2  = (const float*)d_in[16];

    float* out  = (float*)d_out;
    float* seg0 = out;             // x
    float* seg1 = out + BLD;       // x (duplicate)
    float* seg2 = out + 2 * BLD;   // nodes2

    float* ws    = (float*)d_ws;
    float* resid = ws;                 // BLD
    float* x1    = resid + BLD;        // BLD   (layer2 input)
    float* x4    = x1 + BLD;           // BLD
    float* edgeb = x4 + BLD;           // BNED
    float* bigb  = edgeb + BNED;       // BNED  (edge4 / xt2, time-shared)
    float* e1b   = bigb + BNED;        // B*L
    float* a2xb  = e1b + BB * LL;      // B*L
    float* a2eb  = a2xb + BB * LL;     // B*NE
    float* scal  = a2eb + BB * NEE;    // scalars

    const int ML = BB * LL;    // 32768 rows
    const int ME = BB * NEE;   // 65536 rows

    // gather + nodes2 output
    k_gather<<<ML / 8, 256, 0, stream>>>(inputs, emb, emb2, resid, seg2);

    // ---------------- layer 1 (no transfer: xt = x = resid) ----------------
    k_dot128<<<1, 32, 0, stream>>>(g1_q, g1_a, scal);
    k_gemm128<<<ML / 64, 256, 0, stream>>>(resid, g1_w2, x4, ML);
    k_rowdot<<<ML / 8, 256, 0, stream>>>(x4, g1_a + DD, scal, e1b, ML, 1);
    k_edge_agg<<<dim3(BB, 4), 256, 0, stream>>>(resid, HT, e1b, edgeb);
    k_gemm128<<<ME / 64, 256, 0, stream>>>(edgeb, g1_w3, bigb, ME);
    k_rowdot<<<ML / 8, 256, 0, stream>>>(x4, g1_a2, nullptr, a2xb, ML, 0);
    k_rowdot<<<ME / 8, 256, 0, stream>>>(bigb, g1_a2 + DD, nullptr, a2eb, ME, 0);
    k_node_agg<<<dim3(BB, 2), 256, 0, stream>>>(edgeb, HT, a2xb, a2eb,
                                                resid, x1, nullptr);

    // ---------------- layer 2 (transfer: xt = x1 @ g2_w) -------------------
    k_dot128<<<1, 32, 0, stream>>>(g2_q, g2_a, scal + 1);
    k_gemm128<<<ML / 64, 256, 0, stream>>>(x1, g2_w2, x4, ML);
    k_rowdot<<<ML / 8, 256, 0, stream>>>(x4, g2_a + DD, scal + 1, e1b, ML, 1);
    k_gemm128<<<ML / 64, 256, 0, stream>>>(x1, g2_w, bigb, ML);   // xt2
    k_edge_agg<<<dim3(BB, 4), 256, 0, stream>>>(bigb, HT, e1b, edgeb);
    k_gemm128<<<ME / 64, 256, 0, stream>>>(edgeb, g2_w3, bigb, ME); // edge4
    k_rowdot<<<ML / 8, 256, 0, stream>>>(x4, g2_a2, nullptr, a2xb, ML, 0);
    k_rowdot<<<ME / 8, 256, 0, stream>>>(bigb, g2_a2 + DD, nullptr, a2eb, ME, 0);
    k_node_agg<<<dim3(BB, 2), 256, 0, stream>>>(edgeb, HT, a2xb, a2eb,
                                                resid, seg0, seg1);
}
